// RNN_MultiRegional_46583215292498
// MI455X (gfx1250) — matrix operand order repaired and checked by reading, not compiled, not run on hardware
//
#include <hip/hip_runtime.h>
#include <hip/hip_bf16.h>
#include <math.h>

#define HID 1024
#define Q   256
#define BATCH 128
#define TSTEPS 1000
#define NBLK 16            // persistent blocks; block bx owns cols [bx*64, bx*64+64)
#define COLS_PER_BLK 64

typedef __attribute__((ext_vector_type(16))) __bf16 v16bf;
typedef __attribute__((ext_vector_type(8)))  __bf16 v8bf;
typedef __attribute__((ext_vector_type(8)))  float  v8f;

union V16 { v16bf v; v8bf h[2]; };

// ---- d_out layout (floats, concat of reference return tuple) ----
#define OUT_OFF   ((size_t)0)                       // [B,T,1]
#define HN_OFF    ((size_t)128000)                  // [1,B,HID]
#define RNN_OFF   ((size_t)259072)                  // [B,T,HID]
#define XLAST_OFF ((size_t)131331072)               // [1,B,HID]
#define XOUT_OFF  ((size_t)131462144)               // [B,T,HID]

// ---- workspace layout (bytes) ----
#define WS_W      ((size_t)0)                       // W_rec bf16 [HID][HID]   (2 MB)
#define WS_G      ((size_t)2097152)                 // gating bf16 [Q][Q]      (128 KB)
#define WS_H      ((size_t)2228224)                 // h bf16 [2][B][HID]      (512 KB)
#define WS_XTH    ((size_t)2752512)                 // x_thal bf16 [2][B][Q]   (128 KB)
#define WS_X      ((size_t)2883584)                 // x f32 [2][B][HID]       (1 MB)
#define WS_SYNC   ((size_t)3932160)                 // 2 x u32 barrier state

// Build the dense block-structured recurrent matrix once (bf16, row-major [i][j]).
__global__ void build_w_kernel(const float* __restrict__ str2str_fixed,
                               const float* __restrict__ thal2str,
                               const float* __restrict__ alm2str,
                               const float* __restrict__ snr2thal,
                               const float* __restrict__ thal2alm,
                               const float* __restrict__ alm2alm,
                               __bf16* __restrict__ W) {
  int idx = blockIdx.x * blockDim.x + threadIdx.x;
  if (idx >= HID * HID) return;
  int i = idx / HID, j = idx % HID;
  int bi = i / Q, bj = j / Q, ri = i % Q, rj = j % Q;
  float v = 0.0f;
  if (bi == 0) {                                    // striatum rows
    if (bj == 0)      v = -str2str_fixed[ri * Q + rj];
    else if (bj == 2) v = fmaxf(thal2str[ri * Q + rj], 0.0f);
    else if (bj == 3) v = (rj < 52) ? fmaxf(alm2str[ri * Q + rj], 0.0f) : 0.0f;
  } else if (bi == 2) {                             // thalamus rows
    if (bj == 0)      v = fmaxf(snr2thal[ri * Q + rj], 0.0f);
  } else if (bi == 3) {                             // ALM rows
    if (bj == 2)      v = fmaxf(thal2alm[ri * Q + rj], 0.0f);
    else if (bj == 3) v = fmaxf(alm2alm[ri * Q + rj], 0.0f) * ((rj < 180) ? 1.0f : -1.0f);
  }                                                 // bi==1 (SNr rows) all zero
  W[idx] = (__bf16)v;
}

__global__ void build_g_kernel(const float* __restrict__ gating, __bf16* __restrict__ G) {
  int idx = blockIdx.x * blockDim.x + threadIdx.x;
  if (idx >= Q * Q) return;
  G[idx] = (__bf16)gating[idx];
}

__global__ void init_state_kernel(const float* __restrict__ hn, const float* __restrict__ x,
                                  __bf16* __restrict__ h_st, float* __restrict__ x_st,
                                  __bf16* __restrict__ xthal_st, unsigned* __restrict__ syncv) {
  int idx = blockIdx.x * blockDim.x + threadIdx.x;
  if (blockIdx.x == 0 && threadIdx.x < 2) syncv[threadIdx.x] = 0u;  // reset grid barrier
  if (idx >= BATCH * HID) return;
  float xv = x[idx];
  h_st[idx] = (__bf16)hn[idx];
  x_st[idx] = xv;
  int i = idx % HID, b = idx / HID;
  if (i >= 2 * Q && i < 3 * Q) xthal_st[b * Q + (i - 2 * Q)] = (__bf16)xv;
}

// Persistent recurrent scan: one kernel, one device barrier per timestep.
__global__ void __launch_bounds__(256, 1)
rnn_scan_kernel(const float* __restrict__ inp,      // [B][T][4]
                const float* __restrict__ inp_w,    // [4][HID]
                const __bf16* __restrict__ W,       // [HID][HID]
                const __bf16* __restrict__ G,       // [Q][Q]
                __bf16* __restrict__ h_st, __bf16* __restrict__ xthal_st,
                float* __restrict__ x_st, float* __restrict__ d_out,
                unsigned* __restrict__ syncv) {
  const int bx   = blockIdx.x;
  const int tid  = threadIdx.x;
  const int wave = tid >> 5;
  const int lane = tid & 31;
  const int hsel = lane >> 4;       // which half of the wave (K-split / M-split)
  const int lrow = lane & 15;
  const int m0   = wave * 16;       // 8 waves cover all 128 batch rows
  const int nbase = bx * COLS_PER_BLK;
  const int region = nbase / Q;     // 0=str 1=snr 2=thal 3=alm
  const bool do_gate = (region == 0);

  // live K segments per output region (exploit block sparsity of W_rec)
  int s0 = 0, e0 = 0, s1 = 0, e1 = 0;
  if (region == 0) { s0 = 0;   e0 = 256;  s1 = 512; e1 = 832; }   // 832 covers alm2str's 52 cols
  else if (region == 2) { s0 = 0;   e0 = 256; }
  else if (region == 3) { s0 = 512; e0 = 1024; }

  float* rnn_out = d_out + RNN_OFF;
  float* x_out   = d_out + XOUT_OFF;
  float* hn_last = d_out + HN_OFF;
  float* x_last  = d_out + XLAST_OFF;

  // input-projection column weights (constant over time); lane's column per tile
  float wcol[4][4];
  #pragma unroll
  for (int tn = 0; tn < 4; ++tn) {
    int i = nbase + tn * 16 + lrow;
    #pragma unroll
    for (int c = 0; c < 4; ++c) wcol[tn][c] = (i < Q) ? inp_w[c * HID + i] : 0.0f;
  }

  for (int t = 0; t < TSTEPS; ++t) {
    const int cur = t & 1, nxt = cur ^ 1;
    const __bf16* hcur   = h_st     + (size_t)cur * BATCH * HID;
    const __bf16* xthal  = xthal_st + (size_t)cur * BATCH * Q;
    const float*  xcur   = x_st     + (size_t)cur * BATCH * HID;
    __bf16* hnxt    = h_st     + (size_t)nxt * BATCH * HID;
    __bf16* xthal_n = xthal_st + (size_t)nxt * BATCH * Q;
    float*  xnxt    = x_st     + (size_t)nxt * BATCH * HID;

    v8f acc[4]  = {{}, {}, {}, {}};
    v8f gacc[4] = {{}, {}, {}, {}};

    // ---- main recurrent GEMM over live K segments, fp32 acc via bf16 WMMA ----
    for (int seg = 0; seg < 2; ++seg) {
      const int ks = seg ? s1 : s0, ke = seg ? e1 : e0;
      for (int k0 = ks; k0 < ke; k0 += 32) {
        V16 a;
        const __bf16* hrow = hcur + (size_t)(m0 + lrow) * HID + k0 + hsel * 16;
        a.h[0] = *(const v8bf*)(hrow);
        a.h[1] = *(const v8bf*)(hrow + 8);
        #pragma unroll
        for (int tn = 0; tn < 4; ++tn) {
          const __bf16* wrow = W + (size_t)(nbase + tn * 16 + lrow) * HID + k0 + hsel * 16;
          v16bf bfrag = *(const v16bf*)wrow;
          acc[tn] = __builtin_amdgcn_wmma_f32_16x16x32_bf16(
              false, a.v, false, bfrag, (short)0, acc[tn], false, false);
        }
      }
    }

    // ---- thalamic gating GEMM (str columns only): sigmoid(x_thal @ G^T) ----
    if (do_gate) {
      for (int k0 = 0; k0 < Q; k0 += 32) {
        V16 a;
        const __bf16* xrow = xthal + (size_t)(m0 + lrow) * Q + k0 + hsel * 16;
        a.h[0] = *(const v8bf*)(xrow);
        a.h[1] = *(const v8bf*)(xrow + 8);
        #pragma unroll
        for (int tn = 0; tn < 4; ++tn) {
          const __bf16* grow = G + (size_t)(nbase + tn * 16 + lrow) * Q + k0 + hsel * 16;
          v16bf bfrag = *(const v16bf*)grow;
          gacc[tn] = __builtin_amdgcn_wmma_f32_16x16x32_bf16(
              false, a.v, false, bfrag, (short)0, gacc[tn], false, false);
        }
      }
    }

    // per-row raw inputs for this timestep (rows owned by this lane's half)
    float4 inp4[8];
    #pragma unroll
    for (int r = 0; r < 8; ++r) {
      int b = m0 + hsel * 8 + r;
      inp4[r] = *(const float4*)(inp + ((size_t)b * TSTEPS + t) * 4);
    }

    // ---- epilogue: gate, integrate, ReLU, stream out, stage next state ----
    #pragma unroll
    for (int tn = 0; tn < 4; ++tn) {
      const int i = nbase + tn * 16 + lrow;          // output column (uniform region/tile)
      const bool is_str  = (i < Q);
      const bool is_thal = (i >= 2 * Q) && (i < 3 * Q);
      #pragma unroll
      for (int r = 0; r < 8; ++r) {
        const int b = m0 + hsel * 8 + r;             // batch row (C/D layout: VGPR r -> M)
        float u = 0.0f;
        if (is_str)
          u = inp4[r].x * wcol[tn][0] + inp4[r].y * wcol[tn][1] +
              inp4[r].z * wcol[tn][2] + inp4[r].w * wcol[tn][3];
        float gate = 1.0f;
        if (is_str) gate = 1.0f / (1.0f + __expf(-gacc[tn][r]));
        const float xo = xcur[(size_t)b * HID + i];
        const float xn = xo + 0.1f * gate * (acc[tn][r] + u - xo);
        const float hv = fmaxf(xn, 0.0f);
        const size_t oidx = ((size_t)b * TSTEPS + t) * HID + i;
        rnn_out[oidx] = hv;
        x_out[oidx]   = xn;
        hnxt[(size_t)b * HID + i] = (__bf16)hv;
        xnxt[(size_t)b * HID + i] = xn;
        if (is_thal) xthal_n[(size_t)b * Q + (i - 2 * Q)] = (__bf16)xn;
        if (t == TSTEPS - 1) {
          hn_last[(size_t)b * HID + i] = hv;
          x_last[(size_t)b * HID + i]  = xn;
        }
      }
    }

    // ---- device-scope sense-reversing barrier (one per step) ----
    __threadfence();
    __syncthreads();
    if (tid == 0) {
      unsigned gen = __hip_atomic_load(&syncv[1], __ATOMIC_RELAXED, __HIP_MEMORY_SCOPE_AGENT);
      unsigned arrived = __hip_atomic_fetch_add(&syncv[0], 1u, __ATOMIC_ACQ_REL,
                                                __HIP_MEMORY_SCOPE_AGENT);
      if (arrived == NBLK - 1) {
        __hip_atomic_store(&syncv[0], 0u, __ATOMIC_RELAXED, __HIP_MEMORY_SCOPE_AGENT);
        __hip_atomic_store(&syncv[1], gen + 1u, __ATOMIC_RELEASE, __HIP_MEMORY_SCOPE_AGENT);
      } else {
        while (__hip_atomic_load(&syncv[1], __ATOMIC_ACQUIRE, __HIP_MEMORY_SCOPE_AGENT) == gen)
          __builtin_amdgcn_s_sleep(1);
      }
      __threadfence();
    }
    __syncthreads();
  }
}

// Readout: out[b,t] = sigmoid(<h_alm, w_alm> + bias). One wave per (b,t).
__global__ void fc1_kernel(const float* __restrict__ d_out_ro,
                           const float* __restrict__ fc1_w,
                           const float* __restrict__ fc1_b,
                           float* __restrict__ d_out_wr) {
  const int lane  = threadIdx.x & 31;
  const int gwave = blockIdx.x * (blockDim.x >> 5) + (threadIdx.x >> 5);
  if (gwave >= BATCH * TSTEPS) return;
  const int b = gwave / TSTEPS, t = gwave % TSTEPS;
  const float* hrow = d_out_ro + RNN_OFF + ((size_t)b * TSTEPS + t) * HID + 3 * Q;
  const float* wrow = fc1_w + 3 * Q;
  float s = 0.0f;
  #pragma unroll
  for (int k = 0; k < Q / 32; ++k) s += hrow[lane + 32 * k] * wrow[lane + 32 * k];
  #pragma unroll
  for (int off = 16; off > 0; off >>= 1) s += __shfl_down(s, off, 32);
  if (lane == 0)
    d_out_wr[OUT_OFF + (size_t)b * TSTEPS + t] = 1.0f / (1.0f + __expf(-(s + fc1_b[0])));
}

extern "C" void kernel_launch(void* const* d_in, const int* in_sizes, int n_in,
                              void* d_out, int out_size, void* d_ws, size_t ws_size,
                              hipStream_t stream) {
  (void)in_sizes; (void)n_in; (void)out_size; (void)ws_size;
  const float* inp          = (const float*)d_in[0];
  const float* hn           = (const float*)d_in[1];
  const float* x            = (const float*)d_in[2];
  // d_in[3] = str2str_w: multiplied by zeros in the reference -> unused
  const float* thal2alm_w   = (const float*)d_in[4];
  const float* thal2str_w   = (const float*)d_in[5];
  const float* alm2alm_w    = (const float*)d_in[6];
  const float* alm2str_w    = (const float*)d_in[7];
  const float* snr2thal_w   = (const float*)d_in[8];
  const float* thalgating_w = (const float*)d_in[9];
  const float* inp_w        = (const float*)d_in[10];
  const float* str2str_fix  = (const float*)d_in[11];
  const float* fc1_w        = (const float*)d_in[12];
  const float* fc1_b        = (const float*)d_in[13];

  char* ws = (char*)d_ws;
  __bf16*   W        = (__bf16*)(ws + WS_W);
  __bf16*   G        = (__bf16*)(ws + WS_G);
  __bf16*   h_st     = (__bf16*)(ws + WS_H);
  __bf16*   xthal_st = (__bf16*)(ws + WS_XTH);
  float*    x_st     = (float*)(ws + WS_X);
  unsigned* syncv    = (unsigned*)(ws + WS_SYNC);
  float*    out      = (float*)d_out;

  build_w_kernel<<<(HID * HID + 255) / 256, 256, 0, stream>>>(
      str2str_fix, thal2str_w, alm2str_w, snr2thal_w, thal2alm_w, alm2alm_w, W);
  build_g_kernel<<<(Q * Q + 255) / 256, 256, 0, stream>>>(thalgating_w, G);
  init_state_kernel<<<(BATCH * HID + 255) / 256, 256, 0, stream>>>(
      hn, x, h_st, x_st, xthal_st, syncv);
  rnn_scan_kernel<<<NBLK, 256, 0, stream>>>(
      inp, inp_w, W, G, h_st, xthal_st, x_st, out, syncv);
  fc1_kernel<<<(BATCH * TSTEPS) / 8, 256, 0, stream>>>(out, fc1_w, fc1_b, out);
}
